// DualCATANet_11484742550056
// MI455X (gfx1250) — matrix operand order repaired and checked by reading, not compiled, not run on hardware
//
#include <hip/hip_runtime.h>
#include <hip/hip_bf16.h>

#define NN 50000
#define EE 800000
#define DD 128
#define HH 128
#define OUTD 2
#define LL 4
#define AHD 16
#define LN_EPS 1e-5f

typedef __attribute__((ext_vector_type(16))) __bf16 v16bf;
typedef __attribute__((ext_vector_type(8)))  float  v8f;

__device__ __forceinline__ __bf16 f2bf(float f) { return (__bf16)f; }

__device__ __forceinline__ unsigned pack2bf(float a, float b) {
    __bf16 x = (__bf16)a, y = (__bf16)b;
    unsigned short hx, hy;
    __builtin_memcpy(&hx, &x, 2);
    __builtin_memcpy(&hy, &y, 2);
    return (unsigned)hx | ((unsigned)hy << 16);
}

union frag16 { uint4 q[2]; v16bf v; };

// per-layer transposed-weight sizes (bf16 elements)
#define SWST (DD * HH)        // Wst: [col][k]  128x128
#define SWCT (2 * DD * HH)    // Wct: [col][k]  128x256
#define SA1T (AHD * HH)       // A1t: [ai][h]   16x128
#define SPER (SWST + SWCT + SA1T)

// ---------------------------------------------------------------------------
// Zero a float region (agg/deg scratch must be re-zeroed every call/layer).
// ---------------------------------------------------------------------------
__global__ void zero_kernel(float* __restrict__ p, int count) {
    int i = blockIdx.x * blockDim.x + threadIdx.x;
    int stride = gridDim.x * blockDim.x;
    for (; i < count; i += stride) p[i] = 0.0f;
}

// ---------------------------------------------------------------------------
// Pre-convert + transpose all layer weights to bf16 column-major so WMMA
// B-fragments become two contiguous b128 loads per lane (L2/WGP$-resident).
// ---------------------------------------------------------------------------
__global__ void prep_weights_kernel(const float* __restrict__ W_self,  // [L,128,128]
                                    const float* __restrict__ W_ctx,   // [L,256,128]
                                    const float* __restrict__ att_W1,  // [L,128,16]
                                    __bf16* __restrict__ Wst,
                                    __bf16* __restrict__ Wct,
                                    __bf16* __restrict__ A1t) {
    int idx = blockIdx.x * blockDim.x + threadIdx.x;
    if (idx >= LL * SPER) return;
    int l = idx / SPER;
    int r = idx % SPER;
    if (r < SWST) {
        int col = r >> 7, k = r & 127;                  // dst [col][k], K=128
        Wst[l * SWST + r] = f2bf(W_self[l * SWST + k * HH + col]);
    } else if (r < SWST + SWCT) {
        int r2 = r - SWST;
        int col = r2 >> 8, k = r2 & 255;                // dst [col][k], K=256
        Wct[l * SWCT + r2] = f2bf(W_ctx[l * SWCT + k * HH + col]);
    } else {
        int r3 = r - SWST - SWCT;
        int ai = r3 >> 7, h = r3 & 127;                 // dst [ai][h]
        A1t[l * SA1T + r3] = f2bf(att_W1[l * SA1T + h * AHD + ai]);
    }
}

// ---------------------------------------------------------------------------
// Edge aggregation: one wave (32 lanes) per edge, float4 per lane.
//   agg_out[tgt] += x[src], deg_out[tgt] += 1   (aggregate(x, src, tgt))
//   agg_in [src] += x[tgt], deg_in [src] += 1   (aggregate(x, tgt, src))
// HBM-bound phase: coalesced b128 gathers + HW global_atomic_add_f32.
// ---------------------------------------------------------------------------
__global__ void agg_kernel(const float* __restrict__ x,
                           const int* __restrict__ src,
                           const int* __restrict__ tgt,
                           int nedges,
                           float* __restrict__ agg_out,
                           float* __restrict__ agg_in,
                           float* __restrict__ deg_out,
                           float* __restrict__ deg_in) {
    int gid = blockIdx.x * blockDim.x + threadIdx.x;
    int e = gid >> 5;
    int lane = gid & 31;
    if (e >= nedges) return;
    int s = src[e];
    int d = tgt[e];
    if (lane == 0) {
        unsafeAtomicAdd(&deg_out[d], 1.0f);
        unsafeAtomicAdd(&deg_in[s], 1.0f);
    }
    int k = lane * 4;   // 32 lanes x 4 floats = 128 features
    float4 xs = *(const float4*)(x + s * DD + k);
    float4 xd = *(const float4*)(x + d * DD + k);
    float* ao = agg_out + d * DD + k;
    float* ai = agg_in + s * DD + k;
    unsafeAtomicAdd(&ao[0], xs.x);
    unsafeAtomicAdd(&ao[1], xs.y);
    unsafeAtomicAdd(&ao[2], xs.z);
    unsafeAtomicAdd(&ao[3], xs.w);
    unsafeAtomicAdd(&ai[0], xd.x);
    unsafeAtomicAdd(&ai[1], xd.y);
    unsafeAtomicAdd(&ai[2], xd.z);
    unsafeAtomicAdd(&ai[3], xd.w);
}

// ---------------------------------------------------------------------------
// One 16x16 WMMA tile. The full A panel [16 x KTOT] is staged to LDS once
// per view (one barrier pair), composed on-the-fly; then the K-loop is a
// straight run of 2x ds_load_b128 (A) + 2x global_load_b128 (B) + v_wmma.
// ---------------------------------------------------------------------------
template <int KTOT, bool COMPOSE>
__device__ __forceinline__ v8f gemm_tile(const float* __restrict__ xin,
                                         const float* __restrict__ agg,
                                         const float* __restrict__ deg,
                                         const __bf16* __restrict__ Wt, // [128][KTOT]
                                         int row0, int t, int lane, int wv,
                                         __bf16* sA) {
    __syncthreads();   // previous users of sA are done
    // ---- stage A panel, K-half [0,128): x or deg*x - agg ----
#pragma unroll
    for (int p = 0; p < (16 * DD) / 512; ++p) {       // 4 passes, float2/thread
        int i = p * 512 + t * 2;
        int r = i >> 7, kk = i & 127;
        int n = row0 + r;
        float2 val;
        if (!COMPOSE) {
            val = *(const float2*)(xin + n * DD + kk);
        } else {
            float dg = deg[n];
            float2 xv = *(const float2*)(xin + n * DD + kk);
            float2 av = *(const float2*)(agg + n * DD + kk);
            val.x = dg * xv.x - av.x;
            val.y = dg * xv.y - av.y;
        }
        *(unsigned*)((char*)sA + (r * KTOT + kk) * 2) = pack2bf(val.x, val.y);
    }
    // ---- stage A panel, K-half [128,256): agg (composed views only) ----
    if (COMPOSE) {
#pragma unroll
        for (int p = 0; p < (16 * DD) / 512; ++p) {
            int i = p * 512 + t * 2;
            int r = i >> 7, kk = i & 127;
            int n = row0 + r;
            float2 val = *(const float2*)(agg + n * DD + kk);
            *(unsigned*)((char*)sA + (r * KTOT + DD + kk) * 2) = pack2bf(val.x, val.y);
        }
    }
    __syncthreads();

    v8f acc = {};
#pragma unroll
    for (int ks = 0; ks < KTOT; ks += 32) {
        // A fragment: lane<16 row m=lane, K {kA..kA+7, kA+16..kA+23}
        const int m  = lane & 15;
        const int kA = (lane >> 4) * 8;
        frag16 ua;
        const char* pa = (const char*)sA + (m * KTOT + ks + kA) * 2;
        ua.q[0] = *(const uint4*)pa;          // K ks+kA..+7
        ua.q[1] = *(const uint4*)(pa + 32);   // K ks+kA+16..+23

        // B fragment: column-major bf16 weights, K {kB..kB+7, kB+16..kB+23}
        const int col = wv * 16 + (lane & 15);
        const int kB  = (lane >> 4) * 8;
        frag16 ub;
        const __bf16* pb = Wt + col * KTOT + ks + kB;
        ub.q[0] = *(const uint4*)pb;
        ub.q[1] = *(const uint4*)(pb + 16);

        acc = __builtin_amdgcn_wmma_f32_16x16x32_bf16(
            false, ua.v, false, ub.v, (short)0, acc, false, false);
    }
    return acc;
}

// ---------------------------------------------------------------------------
// Fused conv layer: 16 nodes per block, 8 waves, each wave owns a 16-col tile.
// Three WMMA GEMM views -> WMMA attention MLP -> softmax over views ->
// LayerNorm -> ReLU -> residual -> fusion accumulate. One HBM pass of [N,H].
// ---------------------------------------------------------------------------
__global__ void __launch_bounds__(256)
conv_layer_kernel(const float* __restrict__ xin,      // [N,128]
                  const float* __restrict__ agg_out,  // [N,128]
                  const float* __restrict__ agg_in,   // [N,128]
                  const float* __restrict__ deg_out,  // [N]
                  const float* __restrict__ deg_in,   // [N]
                  const __bf16* __restrict__ Wst,     // [128][128] col-major bf16
                  const float* __restrict__ bs,       // [128]
                  const __bf16* __restrict__ Wct,     // [128][256] col-major bf16
                  const float* __restrict__ bc,       // [128]
                  const __bf16* __restrict__ A1t,     // [16][128]  bf16
                  const float* __restrict__ b1,       // [16]
                  const float* __restrict__ a2,       // [16]
                  const float* __restrict__ gamma,    // [128]
                  const float* __restrict__ beta,     // [128]
                  const float* __restrict__ flog,     // [4] fusion logits
                  int layer,
                  float* __restrict__ xout,           // [N,128]
                  float* __restrict__ fused)          // [N,128]
{
    __shared__ float  smv[3][16][HH];   // 3 views x 16 nodes x 128
    __shared__ __bf16 sA[16 * 2 * DD];  // staged A panel (bf16), max K=256
    __shared__ float  sZ[3][16][AHD];
    __shared__ float  sScore[16][3];
    __shared__ float  sWv[16][3];
    __shared__ float  sRed[16][16][2];
    __shared__ float  sMu[16];
    __shared__ float  sRstd[16];

    const int t    = threadIdx.x;
    const int lane = t & 31;
    const int wv   = t >> 5;            // wave id == output col tile (0..7)
    const int row0 = blockIdx.x * 16;

    // prefetch next row-tile of x (predictable stride) into caches
    if (row0 + 16 < NN)
        __builtin_prefetch(xin + (row0 + 16) * DD + lane * 4, 0, 3);

    const int col   = wv * 16 + (lane & 15);
    const int rbase = (lane >> 4) << 3;

    // ---- view 0: self_view = x @ Ws + bs ----
    {
        v8f acc = gemm_tile<DD, false>(xin, xin, deg_out, Wst,
                                       row0, t, lane, wv, sA);
        float bias = bs[col];
#pragma unroll
        for (int j = 0; j < 8; ++j) smv[0][rbase + j][col] = acc[j] + bias;
    }
    // ---- view 1: out_view = [deg_out*x - agg_out, agg_out] @ Wc + bc ----
    {
        v8f acc = gemm_tile<2 * DD, true>(xin, agg_out, deg_out, Wct,
                                          row0, t, lane, wv, sA);
        float bias = bc[col];
#pragma unroll
        for (int j = 0; j < 8; ++j) smv[1][rbase + j][col] = acc[j] + bias;
    }
    // ---- view 2: in_view = [deg_in*x - agg_in, agg_in] @ Wc + bc ----
    {
        v8f acc = gemm_tile<2 * DD, true>(xin, agg_in, deg_in, Wct,
                                          row0, t, lane, wv, sA);
        float bias = bc[col];
#pragma unroll
        for (int j = 0; j < 8; ++j) smv[2][rbase + j][col] = acc[j] + bias;
    }
    __syncthreads();

    // ---- attention MLP via WMMA: Z = mv @ A1 + b1, [16x128]@[128x16] ----
    // All 8 waves execute (EXEC must be all-1s for WMMA); waves 3..7 compute
    // a duplicate view and skip the store.
    {
        const int view = wv % 3;
        v8f acc = {};
#pragma unroll
        for (int ks = 0; ks < DD; ks += 32) {
            const int m  = lane & 15;
            const int kA = (lane >> 4) * 8;
            const float* rowp = &smv[view][m][ks + kA];
            float4 f0 = *(const float4*)(rowp);
            float4 f1 = *(const float4*)(rowp + 4);
            float4 f2 = *(const float4*)(rowp + 16);
            float4 f3 = *(const float4*)(rowp + 20);
            v16bf a;
            a[0] = f2bf(f0.x); a[1] = f2bf(f0.y); a[2] = f2bf(f0.z); a[3] = f2bf(f0.w);
            a[4] = f2bf(f1.x); a[5] = f2bf(f1.y); a[6] = f2bf(f1.z); a[7] = f2bf(f1.w);
            a[8] = f2bf(f2.x); a[9] = f2bf(f2.y); a[10] = f2bf(f2.z); a[11] = f2bf(f2.w);
            a[12] = f2bf(f3.x); a[13] = f2bf(f3.y); a[14] = f2bf(f3.z); a[15] = f2bf(f3.w);

            const int colA = lane & 15;            // attention-hidden index
            const int kB   = (lane >> 4) * 8;
            frag16 ub;
            const __bf16* pb = A1t + colA * DD + ks + kB;
            ub.q[0] = *(const uint4*)pb;
            ub.q[1] = *(const uint4*)(pb + 16);

            acc = __builtin_amdgcn_wmma_f32_16x16x32_bf16(
                false, a, false, ub.v, (short)0, acc, false, false);
        }
        if (wv < 3) {
            const int aiN = lane & 15;
            float bb = b1[aiN];
#pragma unroll
            for (int j = 0; j < 8; ++j)
                sZ[view][rbase + j][aiN] = tanhf(acc[j] + bb);
        }
    }
    __syncthreads();

    // ---- scores, softmax over 3 views ----
    if (t < 48) {
        int v = t % 3, m = t / 3;
        float s = 0.0f;
#pragma unroll
        for (int a = 0; a < AHD; ++a) s += sZ[v][m][a] * a2[a];
        sScore[m][v] = s;
    }
    __syncthreads();
    if (t < 16) {
        float s0 = sScore[t][0], s1 = sScore[t][1], s2 = sScore[t][2];
        float mx = fmaxf(s0, fmaxf(s1, s2));
        float e0 = __expf(s0 - mx), e1 = __expf(s1 - mx), e2 = __expf(s2 - mx);
        float inv = 1.0f / (e0 + e1 + e2);
        sWv[t][0] = e0 * inv; sWv[t][1] = e1 * inv; sWv[t][2] = e2 * inv;
    }
    __syncthreads();

    // ---- combine views, LayerNorm, ReLU, residual, fusion accumulate ----
    {
        const int m = t >> 4;
        const int seg = t & 15;    // 8 columns per thread
        const int c = seg * 8;
        float w0 = sWv[m][0], w1 = sWv[m][1], w2 = sWv[m][2];
        float4 v0a = *(const float4*)&smv[0][m][c];
        float4 v0b = *(const float4*)&smv[0][m][c + 4];
        float4 v1a = *(const float4*)&smv[1][m][c];
        float4 v1b = *(const float4*)&smv[1][m][c + 4];
        float4 v2a = *(const float4*)&smv[2][m][c];
        float4 v2b = *(const float4*)&smv[2][m][c + 4];
        float hv[8];
        hv[0] = w0 * v0a.x + w1 * v1a.x + w2 * v2a.x;
        hv[1] = w0 * v0a.y + w1 * v1a.y + w2 * v2a.y;
        hv[2] = w0 * v0a.z + w1 * v1a.z + w2 * v2a.z;
        hv[3] = w0 * v0a.w + w1 * v1a.w + w2 * v2a.w;
        hv[4] = w0 * v0b.x + w1 * v1b.x + w2 * v2b.x;
        hv[5] = w0 * v0b.y + w1 * v1b.y + w2 * v2b.y;
        hv[6] = w0 * v0b.z + w1 * v1b.z + w2 * v2b.z;
        hv[7] = w0 * v0b.w + w1 * v1b.w + w2 * v2b.w;
        float psum = 0.0f, psq = 0.0f;
#pragma unroll
        for (int j = 0; j < 8; ++j) { psum += hv[j]; psq += hv[j] * hv[j]; }
        sRed[m][seg][0] = psum;
        sRed[m][seg][1] = psq;
        __syncthreads();
        if (t < 16) {
            float s = 0.0f, q = 0.0f;
#pragma unroll
            for (int i = 0; i < 16; ++i) { s += sRed[t][i][0]; q += sRed[t][i][1]; }
            float mu = s * (1.0f / HH);
            float var = q * (1.0f / HH) - mu * mu;
            sMu[t] = mu;
            sRstd[t] = rsqrtf(var + LN_EPS);
        }
        __syncthreads();

        // fusion weight for this layer (softmax over 4 logits, per thread)
        float f0 = flog[0], f1 = flog[1], f2 = flog[2], f3 = flog[3];
        float fmx = fmaxf(fmaxf(f0, f1), fmaxf(f2, f3));
        float ee0 = __expf(f0 - fmx), ee1 = __expf(f1 - fmx);
        float ee2 = __expf(f2 - fmx), ee3 = __expf(f3 - fmx);
        float feinv = 1.0f / (ee0 + ee1 + ee2 + ee3);
        float fw = ((layer == 0) ? ee0 : (layer == 1) ? ee1 : (layer == 2) ? ee2 : ee3) * feinv;

        const int n = row0 + m;
        float mu = sMu[m], rs = sRstd[m];
        float4 xr0 = {0, 0, 0, 0}, xr1 = {0, 0, 0, 0};
        if (layer > 0) {
            xr0 = *(const float4*)(xin + n * DD + c);
            xr1 = *(const float4*)(xin + n * DD + c + 4);
        }
        float xr[8] = {xr0.x, xr0.y, xr0.z, xr0.w, xr1.x, xr1.y, xr1.z, xr1.w};
        float xo[8];
#pragma unroll
        for (int j = 0; j < 8; ++j) {
            float y = (hv[j] - mu) * rs * gamma[c + j] + beta[c + j];
            y = fmaxf(y, 0.0f);                         // ReLU
            xo[j] = y + xr[j];                          // xr==0 for layer 0
        }
        *(float4*)(xout + n * DD + c)     = make_float4(xo[0], xo[1], xo[2], xo[3]);
        *(float4*)(xout + n * DD + c + 4) = make_float4(xo[4], xo[5], xo[6], xo[7]);
        float* fp = fused + n * DD + c;
        if (layer == 0) {
            *(float4*)(fp)     = make_float4(fw * xo[0], fw * xo[1], fw * xo[2], fw * xo[3]);
            *(float4*)(fp + 4) = make_float4(fw * xo[4], fw * xo[5], fw * xo[6], fw * xo[7]);
        } else {
            float4 fa = *(const float4*)(fp);
            float4 fb = *(const float4*)(fp + 4);
            fa.x += fw * xo[0]; fa.y += fw * xo[1]; fa.z += fw * xo[2]; fa.w += fw * xo[3];
            fb.x += fw * xo[4]; fb.y += fw * xo[5]; fb.z += fw * xo[6]; fb.w += fw * xo[7];
            *(float4*)(fp)     = fa;
            *(float4*)(fp + 4) = fb;
        }
    }
}

// ---------------------------------------------------------------------------
// Classifier: out[n,:] = fused[n,:] @ cls_W + cls_b   ([128]x[128,2])
// ---------------------------------------------------------------------------
__global__ void cls_kernel(const float* __restrict__ fused,
                           const float* __restrict__ cW,  // [128,2]
                           const float* __restrict__ cb,  // [2]
                           float* __restrict__ out, int n) {
    int i = blockIdx.x * blockDim.x + threadIdx.x;
    if (i >= n) return;
    float a0 = cb[0], a1 = cb[1];
    const float4* f = (const float4*)(fused + i * HH);
#pragma unroll
    for (int k = 0; k < HH / 4; ++k) {
        float4 v = f[k];
        float4 wa = *(const float4*)(cW + k * 8);
        float4 wb = *(const float4*)(cW + k * 8 + 4);
        a0 += v.x * wa.x + v.y * wa.z + v.z * wb.x + v.w * wb.z;
        a1 += v.x * wa.y + v.y * wa.w + v.z * wb.y + v.w * wb.w;
    }
    out[i * 2 + 0] = a0;
    out[i * 2 + 1] = a1;
}

// ---------------------------------------------------------------------------
extern "C" void kernel_launch(void* const* d_in, const int* in_sizes, int n_in,
                              void* d_out, int out_size, void* d_ws, size_t ws_size,
                              hipStream_t stream) {
    const float* node_features = (const float*)d_in[0];
    const int*   edge_index    = (const int*)d_in[1];
    const float* W_self        = (const float*)d_in[2];   // [L,128,128]
    const float* b_self        = (const float*)d_in[3];   // [L,128]
    const float* W_ctx         = (const float*)d_in[4];   // [L,256,128]
    const float* b_ctx         = (const float*)d_in[5];   // [L,128]
    const float* att_W1        = (const float*)d_in[6];   // [L,128,16]
    const float* att_b1        = (const float*)d_in[7];   // [L,16]
    const float* att_w2        = (const float*)d_in[8];   // [L,16]
    const float* ln_gamma      = (const float*)d_in[9];   // [L,128]
    const float* ln_beta       = (const float*)d_in[10];  // [L,128]
    const float* fusion_logits = (const float*)d_in[11];  // [L]
    const float* cls_W         = (const float*)d_in[12];  // [128,2]
    const float* cls_b         = (const float*)d_in[13];  // [2]
    float* out = (float*)d_out;

    const int* src = edge_index;         // edge_index[0,:]
    const int* tgt = edge_index + EE;    // edge_index[1,:]

    // workspace layout
    float* ws      = (float*)d_ws;
    float* bufA    = ws;                          // [N,128]
    float* bufB    = bufA + NN * HH;              // [N,128]
    float* fusedB  = bufB + NN * HH;              // [N,128]
    float* agg_out = fusedB + NN * HH;            // [N,128]
    float* agg_in  = agg_out + NN * DD;           // [N,128]
    float* deg_out = agg_in + NN * DD;            // [N]
    float* deg_in  = deg_out + NN;                // [N]
    __bf16* Wst = (__bf16*)(deg_in + NN);         // [L][128][128]
    __bf16* Wct = Wst + LL * SWST;                // [L][128][256]
    __bf16* A1t = Wct + LL * SWCT;                // [L][16][128]

    const int zcount = 2 * NN * DD + 2 * NN;      // agg_out..deg_in contiguous
    const int zgrid = 4096;
    const int prep_grid = (LL * SPER + 255) / 256;
    const int agg_grid = (int)(((long long)EE * 32 + 255) / 256);
    const int conv_grid = NN / 16;                // 50000/16 = 3125

    prep_weights_kernel<<<prep_grid, 256, 0, stream>>>(W_self, W_ctx, att_W1,
                                                       Wst, Wct, A1t);

    const float* xin = node_features;
    float* xo = bufA;
    for (int l = 0; l < LL; ++l) {
        zero_kernel<<<zgrid, 256, 0, stream>>>(agg_out, zcount);
        agg_kernel<<<agg_grid, 256, 0, stream>>>(xin, src, tgt, EE,
                                                 agg_out, agg_in, deg_out, deg_in);
        conv_layer_kernel<<<conv_grid, 256, 0, stream>>>(
            xin, agg_out, agg_in, deg_out, deg_in,
            Wst + l * SWST, b_self + l * HH,
            Wct + l * SWCT, b_ctx + l * HH,
            A1t + l * SA1T, att_b1 + l * AHD,
            att_w2 + l * AHD,
            ln_gamma + l * HH, ln_beta + l * HH,
            fusion_logits, l, xo, fusedB);
        xin = xo;
        xo = (xo == bufA) ? bufB : bufA;
    }

    cls_kernel<<<(NN + 255) / 256, 256, 0, stream>>>(fusedB, cls_W, cls_b, out, NN);
}
